// Timeline_13185549598727
// MI455X (gfx1250) — matrix operand order repaired and checked by reading, not compiled, not run on hardware
//
#include <hip/hip_runtime.h>
#include <hip/hip_bf16.h>

typedef __attribute__((ext_vector_type(16))) _Float16 v16h;
typedef __attribute__((ext_vector_type(8)))  float    v8f;

__device__ __forceinline__ float sigm(float x) { return 1.0f / (1.0f + __expf(-x)); }

// ---------------- WMMA fragment helpers (CDNA5 16x16x32 f16 layouts) ----------------
// A matrix 16x32 (MxK), f16: lanes 0-15 rows M=0..15 (K 0-15), lanes 16-31 same rows (K 16-31)
__device__ __forceinline__ v16h load_A(const _Float16* __restrict__ src, int ld, int m0, int k0, int lane) {
    v16h a;
    int m  = lane & 15;
    int kb = (lane & 16) ? 16 : 0;
#pragma unroll
    for (int h = 0; h < 16; ++h) {
        int k = kb + h;
        a[h] = src[(m0 + m) * ld + k0 + k];
    }
    return a;
}

// B matrix 32x16 (KxN) from row-major B[k][n] in LDS
__device__ __forceinline__ v16h load_B(const _Float16* __restrict__ src, int ld, int k0, int n0, int lane) {
    v16h b;
    int col = lane & 15;
    int kb  = (lane & 16) ? 16 : 0;
#pragma unroll
    for (int h = 0; h < 16; ++h)
        b[h] = src[(k0 + kb + h) * ld + n0 + col];
    return b;
}

// B fragment where the source is stored transposed: B[k][n] = src[n][k] (contiguous in h -> ds_load_b128)
__device__ __forceinline__ v16h load_Bt(const _Float16* __restrict__ src, int ld, int k0, int n0, int lane) {
    v16h b;
    int col = lane & 15;
    int kb  = (lane & 16) ? 16 : 0;
#pragma unroll
    for (int h = 0; h < 16; ++h)
        b[h] = src[(n0 + col) * ld + k0 + kb + h];
    return b;
}

// B fragment from fragment-major packed weights: one contiguous 32-byte vector load per lane
__device__ __forceinline__ v16h load_B_packed(const _Float16* __restrict__ src, int NT, int kc, int nt, int lane) {
    return *(const v16h*)(src + (((kc * NT) + nt) * 32 + lane) * 16);
}

__device__ __forceinline__ v8f wmma16(v16h a, v16h b, v8f c) {
    return __builtin_amdgcn_wmma_f32_16x16x32_f16(false, a, false, b, (short)0, c, false, false);
}

// ---------------- Kernel 0: weight f32 -> f16 packed into WMMA B-fragment layout ----------------
// src: (N, 256) row-major W[n][e].  Fragment (kc, nt) holds B[k][n] for e = kc*32+k, n = nt*16+col:
//   dst[ ((kc*NT + nt)*32 + lane)*16 + h ]  with lane = (k&16)|(n&15), h = k&15.
__global__ __launch_bounds__(256) void k_pack_w(const float* __restrict__ src,
                                                _Float16* __restrict__ dst, int N) {
    int idx = blockIdx.x * 256 + threadIdx.x;   // grid = N blocks -> 256*N elems exactly
    int n = idx >> 8;                           // row of W (output feature)
    int e = idx & 255;                          // col of W (input feature / K)
    int kc = e >> 5, k = e & 31;
    int nt = n >> 4, col = n & 15;
    int lane = (k & 16) | col;
    int h    = k & 15;
    int NT   = N >> 4;
    dst[(((kc * NT) + nt) * 32 + lane) * 16 + h] = (_Float16)src[n * 256 + e];
}

// ---------------- Kernel 1: per-visit attention -> vv (B*V, 256) ----------------
__global__ __launch_bounds__(128) void k_attn(const int*   __restrict__ input_seqs,
                                              const float* __restrict__ seq_time_step,
                                              const float* __restrict__ code_masks,
                                              const float* __restrict__ emb,
                                              const _Float16* __restrict__ WQ1p,
                                              const _Float16* __restrict__ WK1p,
                                              const float* __restrict__ decay,
                                              const float* __restrict__ initial,
                                              float* __restrict__ vv) {
    __shared__ _Float16 xh[32 * 256];
    __shared__ _Float16 qh[32 * 256];
    __shared__ _Float16 kh[32 * 256];
    __shared__ float    dpF[32 * 32];
    __shared__ _Float16 spH[32 * 32];
    __shared__ int      sTok[32];
    __shared__ float    sCm[32];
    __shared__ float    sScale[32];

    const int tid  = threadIdx.x;
    const int lane = tid & 31;
    const int wave = tid >> 5;
    const int bv   = blockIdx.x;

    if (tid < 32) sTok[tid] = input_seqs[bv * 32 + tid];
    __syncthreads();
    if (tid < 32) {
        int   tk = sTok[tid];
        float cm = code_masks[bv * 32 + tid];
        float tt = seq_time_step[bv];
        float g  = sigm(decay[tk] * tt + initial[tk]);
        sCm[tid]    = cm;
        sScale[tid] = g * ((1.0e20f - cm) * 1.0e-20f);
    }
    // gather x = emb[tok] as f16
    for (int i = tid; i < 32 * 256; i += 128) {
        int c = i >> 8, d = i & 255;
        xh[i] = (_Float16)emb[(long)sTok[c] * 256 + d];
    }
    __syncthreads();

    // q = x @ WQ1^T, k = x @ WK1^T  (M=32, N=256, K=256); packed-B vector loads
    for (int i = 0; i < 8; ++i) {
        int ti = wave * 8 + i;
        int mt = ti & 1, nt = ti >> 1;
        v8f accq = {}, acck = {};
#pragma unroll 2
        for (int kc = 0; kc < 8; ++kc) {
            v16h a  = load_A(xh, 256, mt * 16, kc * 32, lane);
            v16h bq = load_B_packed(WQ1p, 16, kc, nt, lane);
            v16h bk = load_B_packed(WK1p, 16, kc, nt, lane);
            accq = wmma16(a, bq, accq);
            acck = wmma16(a, bk, acck);
        }
        int col = nt * 16 + (lane & 15);
#pragma unroll
        for (int r = 0; r < 8; ++r) {
            int m = mt * 16 + ((lane & 16) ? 8 : 0) + r;
            qh[m * 256 + col] = (_Float16)accq[r];
            kh[m * 256 + col] = (_Float16)acck[r];
        }
    }
    __syncthreads();

    // dp = q @ k^T  (32x32), one tile per wave
    {
        int mt = wave >> 1, nt = wave & 1;
        v8f acc = {};
#pragma unroll 2
        for (int kc = 0; kc < 8; ++kc) {
            v16h a = load_A(qh, 256, mt * 16, kc * 32, lane);
            v16h b = load_Bt(kh, 256, kc * 32, nt * 16, lane);
            acc = wmma16(a, b, acc);
        }
        int n = nt * 16 + (lane & 15);
#pragma unroll
        for (int r = 0; r < 8; ++r) {
            int m = mt * 16 + ((lane & 16) ? 8 : 0) + r;
            dpF[m * 32 + n] = acc[r];
        }
    }
    __syncthreads();

    // masked softmax over rows (wave32 shuffle reduction)
    for (int r = wave; r < 32; r += 4) {
        float val = (dpF[r * 32 + lane] - sCm[lane] - sCm[r]) * 0.0625f; // 1/sqrt(256)
        float mx = val;
        for (int off = 16; off; off >>= 1) mx = fmaxf(mx, __shfl_xor(mx, off, 32));
        float ex = __expf(val - mx);
        float sm = ex;
        for (int off = 16; off; off >>= 1) sm += __shfl_xor(sm, off, 32);
        spH[r * 32 + lane] = (_Float16)(ex / sm);
    }
    __syncthreads();

    // femb = sp @ x, then vv[d] = sum_c scale[c] * femb[c][d]  (deterministic, no atomics)
    for (int j = 0; j < 4; ++j) {
        int   nt   = wave * 4 + j;
        float csum = 0.0f;
        for (int mt = 0; mt < 2; ++mt) {
            v16h a   = load_A(spH, 32, mt * 16, 0, lane);
            v16h b   = load_B(xh, 256, 0, nt * 16, lane);
            v8f  acc = {};
            acc = wmma16(a, b, acc);
#pragma unroll
            for (int r = 0; r < 8; ++r) {
                int m = mt * 16 + ((lane & 16) ? 8 : 0) + r;
                csum += acc[r] * sScale[m];
            }
        }
        csum += __shfl_xor(csum, 16, 32);
        if (lane < 16) vv[bv * 256 + nt * 16 + lane] = csum;
    }
}

// ---------------- Kernel 2: persistent single-WGP LSTM (64 sequential steps) ----------------
__global__ __launch_bounds__(512) void k_lstm(const float* __restrict__ X, int tStride, int bStride,
                                              const _Float16* __restrict__ Wih_p,
                                              const _Float16* __restrict__ Whh_p,
                                              const float* __restrict__ bias,
                                              float* __restrict__ zbuf,
                                              float* __restrict__ out /* (64,32,256) time-major */) {
    __shared__ _Float16 xtH[32 * 256];
    __shared__ _Float16 hH[32 * 256];
    __shared__ float    cF[32 * 256];

    const int tid  = threadIdx.x;
    const int lane = tid & 31;
    const int wave = tid >> 5;

    for (int i = tid; i < 32 * 256; i += 512) {
        hH[i] = (_Float16)0.0f;
        cF[i] = 0.0f;
    }
    __syncthreads();

    for (int t = 0; t < 64; ++t) {
        // stage x_t
        for (int i = tid; i < 32 * 256; i += 512) {
            int m = i >> 8, d = i & 255;
            xtH[i] = (_Float16)X[m * bStride + t * tStride + d];
        }
        __syncthreads();

        // z = x_t @ Wih^T + h @ Whh^T + b   (M=32, N=1024, K=256 twice) — 128 tiles over 16 waves
        for (int i = 0; i < 8; ++i) {
            int ti = wave * 8 + i;
            int mt = ti & 1, nt = ti >> 1;
            __builtin_prefetch(Wih_p + (long)nt * 512, 0, 1);
            v8f acc = {};
#pragma unroll 2
            for (int kc = 0; kc < 8; ++kc) {
                v16h a = load_A(xtH, 256, mt * 16, kc * 32, lane);
                v16h b = load_B_packed(Wih_p, 64, kc, nt, lane);
                acc = wmma16(a, b, acc);
            }
#pragma unroll 2
            for (int kc = 0; kc < 8; ++kc) {
                v16h a = load_A(hH, 256, mt * 16, kc * 32, lane);
                v16h b = load_B_packed(Whh_p, 64, kc, nt, lane);
                acc = wmma16(a, b, acc);
            }
            int   n  = nt * 16 + (lane & 15);
            float bb = bias[n];
#pragma unroll
            for (int r = 0; r < 8; ++r) {
                int m = mt * 16 + ((lane & 16) ? 8 : 0) + r;
                zbuf[m * 1024 + n] = acc[r] + bb;
            }
        }
        // make the z stores visible block-wide before the barrier (CDNA5 split counters)
        asm volatile("s_wait_storecnt 0x0" ::: "memory");
        __syncthreads();

        // gates + state update
        for (int i = tid; i < 32 * 256; i += 512) {
            int   m = i >> 8, d = i & 255;
            float zi = zbuf[m * 1024 + d];
            float zf = zbuf[m * 1024 + 256 + d];
            float zg = zbuf[m * 1024 + 512 + d];
            float zo = zbuf[m * 1024 + 768 + d];
            float cn = sigm(zf) * cF[i] + sigm(zi) * tanhf(zg);
            float hv = sigm(zo) * tanhf(cn);
            cF[i] = cn;
            hH[i] = (_Float16)hv;
            out[(t * 32 + m) * 256 + d] = hv;
        }
        __syncthreads();
    }
}

// ---------------- Kernel 3: tail (only v* = lengths[b]-1 matters for the output) ----------------
__global__ __launch_bounds__(256) void k_tail(const int* __restrict__ lengths,
                                              const int* __restrict__ diff_t,
                                              const float* __restrict__ betas,
                                              const float* __restrict__ lout1, /* t_lstm time-major */
                                              const float* __restrict__ hs,    /* hidden time-major */
                                              const float* __restrict__ whk_W, const float* __restrict__ whk_b,
                                              const float* __restrict__ w1_W,  const float* __restrict__ w1_b,
                                              const float* __restrict__ w2_W,  const float* __restrict__ w2_b,
                                              const float* __restrict__ Wdiff, const float* __restrict__ noise,
                                              const float* __restrict__ fuse_W, const float* __restrict__ fuse_b,
                                              const float* __restrict__ lab_W, const float* __restrict__ lab_b,
                                              float* __restrict__ out) {
    __shared__ float sEk[256], sWhp[256], sNoisy[256], sGen[256], sTl[256], sU[64], red[256];
    __shared__ float sAlpha, sA0, sA1;

    const int tid   = threadIdx.x;
    const int b     = blockIdx.x;
    const int vstar = lengths[b] - 1;

    if (tid == 0) {
        int   tt = diff_t[b];
        float a  = 1.0f;
        for (int i = 0; i <= tt; ++i) a *= (1.0f - betas[i]);
        sAlpha = a;
    }
    sEk[tid] = lout1[b * 256 + tid];                      // t_lstm[b,0,:]
    sTl[tid] = lout1[(vstar * 32 + b) * 256 + tid];       // t_lstm[b,v*,:]
    __syncthreads();

    float alignedv;
    if (vstar == 0) {
        alignedv = sEk[tid];
    } else {
        const float* hrow = hs + ((vstar - 1) * 32 + b) * 256;
        float acc = whk_b[tid];
        for (int e = 0; e < 256; ++e) acc += hrow[e] * whk_W[tid * 256 + e];
        sWhp[tid] = acc;
        __syncthreads();
        if (tid < 64) {
            float        a2 = w1_b[tid];
            const float* wr = w1_W + tid * 512;
            for (int e = 0; e < 256; ++e) a2 += sEk[e] * wr[e];
            for (int e = 0; e < 256; ++e) a2 += sWhp[e] * wr[256 + e];
            sU[tid] = tanhf(a2);
        }
        __syncthreads();
        if (tid == 0) {
            float l0 = w2_b[0], l1 = w2_b[1];
            for (int h = 0; h < 64; ++h) { l0 += sU[h] * w2_W[h]; l1 += sU[h] * w2_W[64 + h]; }
            float mx = fmaxf(l0, l1);
            float e0 = __expf(l0 - mx), e1 = __expf(l1 - mx);
            sA0 = e0 / (e0 + e1);
            sA1 = e1 / (e0 + e1);
        }
        __syncthreads();
        alignedv = sEk[tid] * sA0 + sWhp[tid] * sA1;
    }

    float nz = noise[(b * 64 + vstar) * 256 + tid];
    float sa = sqrtf(sAlpha);
    float sb = sqrtf(fmaxf(0.0f, 1.0f - sAlpha));
    sNoisy[tid] = alignedv * sa + nz * sb;
    __syncthreads();

    float pn = 0.0f;
    for (int e = 0; e < 256; ++e) pn += sNoisy[e] * Wdiff[e * 256 + tid];
    sGen[tid] = alignedv + nz - pn;
    __syncthreads();

    float        f  = fuse_b[tid];
    const float* fr = fuse_W + tid * 512;
    for (int e = 0; e < 256; ++e) f += sTl[e] * fr[e];
    for (int e = 0; e < 256; ++e) f += sGen[e] * fr[256 + e];

    for (int k = 0; k < 2; ++k) {
        red[tid] = f * lab_W[k * 256 + tid];
        __syncthreads();
        for (int s = 128; s > 0; s >>= 1) {
            if (tid < s) red[tid] += red[tid + s];
            __syncthreads();
        }
        if (tid == 0) out[b * 2 + k] = red[0] + lab_b[k];
        __syncthreads();
    }
}

extern "C" void kernel_launch(void* const* d_in, const int* in_sizes, int n_in,
                              void* d_out, int out_size, void* d_ws, size_t ws_size,
                              hipStream_t stream) {
    (void)in_sizes; (void)n_in; (void)out_size; (void)ws_size;

    const int*   input_seqs = (const int*)d_in[0];
    const int*   lengths    = (const int*)d_in[2];
    const float* seq_time   = (const float*)d_in[3];
    const float* code_masks = (const float*)d_in[4];
    const float* emb        = (const float*)d_in[5];
    const float* WQ1        = (const float*)d_in[6];
    const float* WK1        = (const float*)d_in[7];
    const float* decay      = (const float*)d_in[8];
    const float* initial    = (const float*)d_in[9];
    const float* lstm_Wih   = (const float*)d_in[10];
    const float* lstm_Whh   = (const float*)d_in[11];
    const float* lstm_b     = (const float*)d_in[12];
    const float* hs_Wih     = (const float*)d_in[13];
    const float* hs_Whh     = (const float*)d_in[14];
    const float* hs_b       = (const float*)d_in[15];
    const float* whk_W      = (const float*)d_in[16];
    const float* whk_b      = (const float*)d_in[17];
    const float* w1_W       = (const float*)d_in[18];
    const float* w1_b       = (const float*)d_in[19];
    const float* w2_W       = (const float*)d_in[20];
    const float* w2_b       = (const float*)d_in[21];
    const float* fuse_W     = (const float*)d_in[22];
    const float* fuse_b     = (const float*)d_in[23];
    const float* lab_W      = (const float*)d_in[24];
    const float* lab_b      = (const float*)d_in[25];
    const float* betas      = (const float*)d_in[26];
    const float* Wdiff      = (const float*)d_in[27];
    const int*   diff_t     = (const int*)d_in[28];
    const float* noise      = (const float*)d_in[29];

    char* ws = (char*)d_ws;
    _Float16* WQ1p  = (_Float16*)(ws + 0);         // 256x256 f16 packed (128 KB)
    _Float16* WK1p  = (_Float16*)(ws + 131072);    // 128 KB
    _Float16* Wih1p = (_Float16*)(ws + 262144);    // 256x1024 f16 packed (512 KB)
    _Float16* Whh1p = (_Float16*)(ws + 786432);
    _Float16* Wih2p = (_Float16*)(ws + 1310720);
    _Float16* Whh2p = (_Float16*)(ws + 1835008);
    float*    vv    = (float*)(ws + 2359296);      // (2048,256) f32, 2 MB
    float*    lout1 = (float*)(ws + 4456448);      // (64,32,256) f32, 2 MB
    float*    lout2 = (float*)(ws + 6553600);      // (64,32,256) f32, 2 MB
    float*    zbuf  = (float*)(ws + 8650752);      // (32,1024) f32, 128 KB

    // weight prep (f32 -> f16, packed into WMMA B-fragment layout)
    k_pack_w<<<256,  256, 0, stream>>>(WQ1,      WQ1p,  256);
    k_pack_w<<<256,  256, 0, stream>>>(WK1,      WK1p,  256);
    k_pack_w<<<1024, 256, 0, stream>>>(lstm_Wih, Wih1p, 1024);
    k_pack_w<<<1024, 256, 0, stream>>>(lstm_Whh, Whh1p, 1024);
    k_pack_w<<<1024, 256, 0, stream>>>(hs_Wih,   Wih2p, 1024);
    k_pack_w<<<1024, 256, 0, stream>>>(hs_Whh,   Whh2p, 1024);

    // per-visit attention
    k_attn<<<2048, 128, 0, stream>>>(input_seqs, seq_time, code_masks, emb,
                                     WQ1p, WK1p, decay, initial, vv);

    // LSTM 1: input vv laid out (B,V,D) -> bStride=64*256, tStride=256
    k_lstm<<<1, 512, 0, stream>>>(vv, 256, 16384, Wih1p, Whh1p, lstm_b, zbuf, lout1);
    // LSTM 2: input lout1 time-major (V,B,D) -> tStride=32*256, bStride=256
    k_lstm<<<1, 512, 0, stream>>>(lout1, 8192, 256, Wih2p, Whh2p, hs_b, zbuf, lout2);

    // tail: attention-align, diffusion, fuse, classify — only row v* matters
    k_tail<<<32, 256, 0, stream>>>(lengths, diff_t, betas, lout1, lout2,
                                   whk_W, whk_b, w1_W, w1_b, w2_W, w2_b,
                                   Wdiff, noise, fuse_W, fuse_b, lab_W, lab_b,
                                   (float*)d_out);
}
